// DeformableConv2d_6073083756680
// MI455X (gfx1250) — matrix operand order repaired and checked by reading, not compiled, not run on hardware
//
#include <hip/hip_runtime.h>

typedef _Float16 v16h __attribute__((ext_vector_type(16)));
typedef _Float16 v8h  __attribute__((ext_vector_type(8)));
typedef _Float16 v4h  __attribute__((ext_vector_type(4)));
typedef float    v8f  __attribute__((ext_vector_type(8)));

#define C_       128
#define F_       256
#define N_       9
#define K_       1152        // N_ * C_
#define KC_      36          // K_ / 32
#define MT_      64          // pixels per workgroup
#define LDSPITCH 1160        // K_ + 8 halves padding
#define NPIX     (8 * 64 * 64)

// ---------------------------------------------------------------------------
// Kernel 1: pack W[(n*128+c), f] fp32 -> f16 WMMA B-fragment blocks in d_ws.
// Block (kc, ft) is a 32(K) x 16(F) tile, 512 halves; lane L's 16 halves are
// contiguous:  lane 0-15: N=L, K=kc*32+{0..7,16..23} ; lane 16-31: N=L-16, K=+{8..15,24..31}
// ---------------------------------------------------------------------------
__global__ __launch_bounds__(256) void pack_w(const float* __restrict__ W,
                                              _Float16* __restrict__ wsB) {
  int e      = blockIdx.x * 256 + threadIdx.x;   // 0 .. K_*F_-1 (294912)
  int blk    = e >> 9;                           // kc*16 + ft
  int within = e & 511;
  int lane   = within >> 4;
  int j      = within & 15;
  int kc = blk >> 4;
  int ft = blk & 15;
  int k = kc * 32 + ((j & 8) << 1) + ((lane & 16) >> 1) + (j & 7);
  int f = ft * 16 + (lane & 15);
  wsB[e] = (_Float16)W[(size_t)k * F_ + f];
}

// ---------------------------------------------------------------------------
// Kernel 2: fused deformable bilinear sampling (fp32 -> f16 LDS A-panel)
//           + f16 WMMA GEMM, f32 accumulate, bias.
// One WG = 64 pixels x 256 features. Wave w owns F-cols [32w, 32w+32) across
// all four 16-row M-subtiles: each B fragment is loaded from L2 exactly once
// per WG and reused 4x in registers; B is double-buffered across the K loop.
// ---------------------------------------------------------------------------
__global__ __launch_bounds__(256) void deform_wmma(const float* __restrict__ x,
                                                   const float* __restrict__ offs,
                                                   const _Float16* __restrict__ wsB,
                                                   const float* __restrict__ bias,
                                                   float* __restrict__ out) {
  extern __shared__ _Float16 As[];   // MT_ * LDSPITCH halves = 148480 bytes

  const int tid  = threadIdx.x;
  const int wid  = tid >> 5;         // wave32: 8 waves per WG
  const int lane = tid & 31;
  const int p0   = blockIdx.x * MT_;

  // Faithful replication of the reference's quirky (2,3,3)->(9,2) reshape:
  const int iy [9] = {0, 0, 1, 2, 2, 1, 0, 2, 1};
  const int ixt[9] = {0, 1, 1, 2, 0, 2, 1, 0, 2};

  // ---------------- phase 1: bilinear sampling into LDS -------------------
  for (int t = wid; t < MT_ * N_; t += 8) {
    int pl = t / N_;
    int n  = t - pl * N_;
    int p  = p0 + pl;
    int b  = p >> 12;
    int rm = p & 4095;
    int yy = rm >> 6;
    int xx = rm & 63;

    const float* op = offs + (size_t)p * (2 * N_) + 2 * n;
    float offy = op[0];
    float offx = op[1];
    float cy = fminf(fmaxf((float)(yy - 1 + iy [n]) + offy, 0.f), 63.f);
    float cx = fminf(fmaxf((float)(xx - 1 + ixt[n]) + offx, 0.f), 63.f);
    float y0f = floorf(cy), x0f = floorf(cx);
    int y0 = (int)y0f,      x0 = (int)x0f;
    int y1 = (int)ceilf(cy), x1 = (int)ceilf(cx);
    float fy = cy - y0f, fx = cx - x0f;

    const float* xb = x + (size_t)b * (4096 * C_);
    const float4 vlt = ((const float4*)(xb + (y0 * 64 + x0) * C_))[lane];
    const float4 vrb = ((const float4*)(xb + (y1 * 64 + x1) * C_))[lane];
    const float4 vlb = ((const float4*)(xb + (y0 * 64 + x1) * C_))[lane]; // (y0,x1)
    const float4 vrt = ((const float4*)(xb + (y1 * 64 + x0) * C_))[lane]; // (y1,x0)

    v4h hv;
#define BILIN(i, comp)                                            \
    {                                                             \
      float vt_ = fmaf(fy, vrt.comp - vlt.comp, vlt.comp);        \
      float vb_ = fmaf(fy, vrb.comp - vlb.comp, vlb.comp);        \
      hv[i] = (_Float16)fmaf(fx, vb_ - vt_, vt_);                 \
    }
    BILIN(0, x) BILIN(1, y) BILIN(2, z) BILIN(3, w)
#undef BILIN
    *((v4h*)&As[pl * LDSPITCH + n * C_ + lane * 4]) = hv;   // ds_store_b64
  }
  __syncthreads();

  // ---------------- phase 2: WMMA GEMM ------------------------------------
  // A frag (16-bit A layout): lane<16 -> K {0..7,16..23}; lane>=16 -> {8..15,24..31}
  const int khalf = (lane & 16) ? 8 : 0;
  const _Float16* arow0 = &As[(lane & 15) * LDSPITCH + khalf];
  const _Float16* bbase = wsB + (size_t)(wid * 2) * 512 + lane * 16;

  v8f acc[8] = {};   // [m*2 + t]

#define LOADB(kc, t) (*(const v16h*)(bbase + (size_t)(kc) * (16 * 512) + (t) * 512))
#define LOADA(kc, m)                                                          \
  __builtin_shufflevector(                                                    \
      *(const v8h*)(arow0 + (m) * (16 * LDSPITCH) + (kc) * 32),               \
      *(const v8h*)(arow0 + (m) * (16 * LDSPITCH) + (kc) * 32 + 16),          \
      0, 1, 2, 3, 4, 5, 6, 7, 8, 9, 10, 11, 12, 13, 14, 15)

  v16h b0 = LOADB(0, 0);
  v16h b1 = LOADB(0, 1);

  for (int kc = 0; kc < KC_; ++kc) {
    // prefetch next chunk's B fragments (double buffer)
    int kn = (kc + 1 < KC_) ? kc + 1 : kc;
    v16h nb0 = LOADB(kn, 0);
    v16h nb1 = LOADB(kn, 1);
#pragma unroll
    for (int m = 0; m < 4; ++m) {
      v16h a = LOADA(kc, m);
      acc[m * 2 + 0] = __builtin_amdgcn_wmma_f32_16x16x32_f16(
          false, a, false, b0, (short)0, acc[m * 2 + 0], false, false);
      acc[m * 2 + 1] = __builtin_amdgcn_wmma_f32_16x16x32_f16(
          false, a, false, b1, (short)0, acc[m * 2 + 1], false, false);
    }
    b0 = nb0;
    b1 = nb1;
  }
#undef LOADA
#undef LOADB

  // ---------------- epilogue: C/D layout + bias ---------------------------
  // VGPR r: lanes 0-15 -> M=r, N=lane; lanes 16-31 -> M=r+8, N=lane-16
  const int m_hi = (lane & 16) ? 8 : 0;
#pragma unroll
  for (int t = 0; t < 2; ++t) {
    int fcol = wid * 32 + t * 16 + (lane & 15);
    float bv = bias[fcol];
#pragma unroll
    for (int m = 0; m < 4; ++m) {
      int prow = p0 + m * 16 + m_hi;
#pragma unroll
      for (int r = 0; r < 8; ++r) {
        out[(size_t)(prow + r) * F_ + fcol] = acc[m * 2 + t][r] + bv;
      }
    }
  }
}

// ---------------------------------------------------------------------------
extern "C" void kernel_launch(void* const* d_in, const int* in_sizes, int n_in,
                              void* d_out, int out_size, void* d_ws, size_t ws_size,
                              hipStream_t stream) {
  const float* x    = (const float*)d_in[0];   // (8,64,64,128) fp32
  const float* offs = (const float*)d_in[1];   // (8,64,64,18)  fp32
  const float* W    = (const float*)d_in[2];   // (9,128,256)   fp32
  const float* b    = (const float*)d_in[3];   // (256,)        fp32
  float* out        = (float*)d_out;           // (8,64,64,256) fp32
  _Float16* wsB     = (_Float16*)d_ws;         // 576 KB packed f16 W

  pack_w<<<(K_ * F_) / 256, 256, 0, stream>>>(W, wsB);
  deform_wmma<<<NPIX / MT_, 256, MT_ * LDSPITCH * sizeof(_Float16), stream>>>(
      x, offs, wsB, b, out);
}
